// MultiHeadAttention_78245714198674
// MI455X (gfx1250) — compile-verified, matrix operand order checked by reading, not run on hardware
//
#include <hip/hip_runtime.h>
#include <hip/hip_bf16.h>

// MI455X / gfx1250 multi-head attention, bf16 WMMA with f32 accumulation.
// Flash-style streaming softmax (no S x S materialization).
// GEMMs stage A/W tiles into LDS via GLOBAL_LOAD_ASYNC_TO_LDS_B128 with
// double buffering (ASYNCcnt), fragments consumed from LDS via ds_load_b128.
//
// Workspace layout (bf16): x, wq, wk, wv, wo converted; Q, K row-major
// [B*S, D]; V transposed [B,H,DH,S]; context [B*S, D]. ~92.3 MB total.

typedef __attribute__((ext_vector_type(16))) __bf16 v16bf;
typedef __attribute__((ext_vector_type(8)))  __bf16 v8bf;
typedef __attribute__((ext_vector_type(4)))  __bf16 v4bf;
typedef __attribute__((ext_vector_type(8)))  float  v8f;

constexpr int Bc = 4, Sc = 2048, Dc = 1024, Hc = 16, DHc = 64;
constexpr int Mtok = Bc * Sc;  // 8192 tokens

union Frag16 { v16bf v; v8bf h[2]; };

// Load a 16(row) x 32(col) bf16 fragment where the register-lane-major
// dimension ("row") is contiguous in memory with leading dim `ld`.
// Matches the documented 16-bit A layout (lanes 0-15: rows, K 0..7/16..23;
// lanes 16-31: same rows, K 8..15/24..31); B-side operands are arranged in
// memory (W row-major, K rows, V pre-transposed) so the same loader applies.
static __device__ __forceinline__ v16bf load_frag(const __bf16* base, int ld) {
  const int lane = threadIdx.x & 31;
  const int r  = lane & 15;
  const int kb = (lane >> 4) * 8;
  Frag16 f;
  f.h[0] = *reinterpret_cast<const v8bf*>(base + (size_t)r * ld + kb);
  f.h[1] = *reinterpret_cast<const v8bf*>(base + (size_t)r * ld + kb + 16);
  return f.v;
}

static __device__ __forceinline__ v8f wmma_bf16(v16bf a, v16bf b, v8f c) {
  // D = A(16x32) x B(32x16) + C, f32 accum
  return __builtin_amdgcn_wmma_f32_16x16x32_bf16(
      /*neg_a=*/false, a, /*neg_b=*/false, b,
      /*c_mod=*/(short)0, c, /*reuse_a=*/false, /*reuse_b=*/false);
}

// Fire-and-forget 16B-per-lane copy global -> LDS, tracked by ASYNCcnt.
static __device__ __forceinline__ void async_copy_b128(const __bf16* gsrc,
                                                       const __bf16* ldst) {
  const unsigned long long ga = (unsigned long long)(uintptr_t)gsrc;
  const unsigned int la = (unsigned int)(uintptr_t)ldst;  // LDS = addr[31:0]
  asm volatile("global_load_async_to_lds_b128 %0, %1, off"
               :: "v"(la), "v"(ga) : "memory");
}
static __device__ __forceinline__ void wait_async0() {
  asm volatile("s_wait_asynccnt 0" ::: "memory");
}

// Reductions across the 16-lane half-wave (C/D layout: one matrix row lives
// on 16 lanes; xor masks 1,2,4,8 stay within the half).
static __device__ __forceinline__ float rmax16(float v) {
#pragma unroll
  for (int m = 1; m < 16; m <<= 1) v = fmaxf(v, __shfl_xor(v, m, 32));
  return v;
}
static __device__ __forceinline__ float rsum16(float v) {
#pragma unroll
  for (int m = 1; m < 16; m <<= 1) v += __shfl_xor(v, m, 32);
  return v;
}

// ---------------------------------------------------------------- convert
__global__ void __launch_bounds__(256) cvt_f32_bf16(
    const float* __restrict__ in, __bf16* __restrict__ out, int n4) {
  int i = blockIdx.x * blockDim.x + threadIdx.x;
  if (i >= n4) return;
  float4 f = reinterpret_cast<const float4*>(in)[i];
  v4bf r;
  r[0] = (__bf16)f.x; r[1] = (__bf16)f.y; r[2] = (__bf16)f.z; r[3] = (__bf16)f.w;
  reinterpret_cast<v4bf*>(out)[i] = r;
}

// ---------------------------------------------------------------- GEMM
// y[m,n] = sum_k A[m,k] * W[n,k] + bias[n]   (torch Linear: x @ W^T + b)
// Block: 256 threads = 8 waves, block tile 128(M) x 128(N), K stepped by 32.
// A/W 128x32 tiles double-buffered in LDS, filled by async-to-LDS copies
// overlapping the previous step's WMMAs. Wave tile 32x64 = 2x4 accums.
// MODE 0: bf16 row-major out. MODE 1: bf16 out transposed to [B,H,DH,S].
// MODE 2: f32 row-major out (final projection).
template <int MODE>
__global__ void __launch_bounds__(256) gemm_wmma_k(
    const __bf16* __restrict__ A, const __bf16* __restrict__ W,
    const float* __restrict__ bias, __bf16* __restrict__ outb,
    float* __restrict__ outf) {
  __shared__ __align__(16) __bf16 Asl[2][128 * 32];
  __shared__ __align__(16) __bf16 Wsl[2][128 * 32];
  const int tid  = threadIdx.x;
  const int wave = tid >> 5;
  const int wm = wave & 3;                   // 4 waves along M
  const int wn = wave >> 2;                  // 2 waves along N
  const int m0b = blockIdx.y * 128;
  const int n0b = blockIdx.x * 128;

  // Stage one 128x32 A tile + one 128x32 W tile into buffer `buf`.
  // 512 16B-chunks per tile; 256 threads -> 2 chunks each per tile.
  auto stage = [&](int buf, int k) {
#pragma unroll
    for (int i = 0; i < 2; ++i) {
      const int c = tid + i * 256;           // chunk id 0..511
      const int row = c >> 2;
      const int col = (c & 3) * 8;           // bf16 elements
      async_copy_b128(A + (size_t)(m0b + row) * Dc + k + col,
                      &Asl[buf][row * 32 + col]);
      async_copy_b128(W + (size_t)(n0b + row) * Dc + k + col,
                      &Wsl[buf][row * 32 + col]);
    }
  };

  stage(0, 0);

  v8f acc[2][4];
#pragma unroll
  for (int i = 0; i < 2; ++i)
#pragma unroll
    for (int j = 0; j < 4; ++j) acc[i][j] = (v8f){};

  int buf = 0;
  for (int k = 0; k < Dc; k += 32, buf ^= 1) {
    wait_async0();       // this wave's copies into `buf` are in LDS
    __syncthreads();     // everyone's copies into `buf` are in LDS
    if (k + 32 < Dc) stage(buf ^ 1, k + 32);  // overlap next tile copy

    const __bf16* Ab = &Asl[buf][(wm * 32) * 32];
    const __bf16* Wb = &Wsl[buf][(wn * 64) * 32];
    v16bf af[2], bfm[4];
    af[0] = load_frag(Ab, 32);
    af[1] = load_frag(Ab + 16 * 32, 32);
#pragma unroll
    for (int j = 0; j < 4; ++j) bfm[j] = load_frag(Wb + j * 16 * 32, 32);
#pragma unroll
    for (int i = 0; i < 2; ++i)
#pragma unroll
      for (int j = 0; j < 4; ++j) acc[i][j] = wmma_bf16(af[i], bfm[j], acc[i][j]);
    // no trailing barrier: `buf` is next overwritten only after the
    // wait+barrier of the following iteration (double buffer).
  }

  const int lane = threadIdx.x & 31;
  const int nl = lane & 15;            // output column within 16-tile
  const int mb = (lane >> 4) * 8;      // rows r (+0 or +8) per lane half
#pragma unroll
  for (int i = 0; i < 2; ++i) {
#pragma unroll
    for (int j = 0; j < 4; ++j) {
      const int ncol = n0b + wn * 64 + j * 16 + nl;
      const float bval = bias[ncol];
#pragma unroll
      for (int r = 0; r < 8; ++r) {
        const int mrow = m0b + wm * 32 + i * 16 + mb + r;
        const float val = acc[i][j][r] + bval;
        if (MODE == 0) {
          outb[(size_t)mrow * Dc + ncol] = (__bf16)val;
        } else if (MODE == 1) {
          const int bb = mrow >> 11, s = mrow & (Sc - 1);
          const int hh = ncol >> 6, dh = ncol & (DHc - 1);
          outb[(((size_t)bb * Hc + hh) * DHc + dh) * Sc + s] = (__bf16)val;
        } else {
          outf[(size_t)mrow * Dc + ncol] = val;
        }
      }
    }
  }
}

// ---------------------------------------------------------------- attention
// One wave per (b, h, 16-row q tile). Streams keys in chunks of 32:
// 4 WMMAs for S = Q K^T (two 16-key tiles x two K=32 steps over DH=64),
// online softmax with shuffle row-stats, P bounced bf16 through per-wave LDS
// into A-fragment layout, 4 WMMAs for P @ V (V is pre-transposed [B,H,DH,S]).
__global__ void __launch_bounds__(256) attn_kernel(
    const __bf16* __restrict__ Q, const __bf16* __restrict__ K,
    const __bf16* __restrict__ V,  // [B,H,DH,S]
    __bf16* __restrict__ C) {
  __shared__ __align__(16) __bf16 Pl[8][16 * 32];
  const int wave = threadIdx.x >> 5;
  const int lane = threadIdx.x & 31;
  const int flat = blockIdx.x * 8 + wave;      // 8192 wave-tiles
  const int qt = flat & 127;                   // S/16 tiles
  const int h  = (flat >> 7) & 15;
  const int b  = flat >> 11;
  const int q0 = qt << 4;
  const float scale = 0.125f;                  // 1/sqrt(64)

  const size_t qoff = ((size_t)b * Sc + q0) * Dc + (size_t)h * DHc;
  const v16bf qf0 = load_frag(Q + qoff, Dc);
  const v16bf qf1 = load_frag(Q + qoff + 32, Dc);

  float mi[8], li[8];
  v8f co[4];
#pragma unroll
  for (int r = 0; r < 8; ++r) { mi[r] = -3.0e30f; li[r] = 0.0f; }
#pragma unroll
  for (int j = 0; j < 4; ++j) co[j] = (v8f){};

  const int nl = lane & 15;
  const int mb = (lane >> 4) * 8;
  __bf16* P = &Pl[wave][0];

  for (int kk = 0; kk < Sc; kk += 32) {
    const size_t kof = ((size_t)b * Sc + kk) * Dc + (size_t)h * DHc;
    v8f s0 = (v8f){}, s1 = (v8f){};
    s0 = wmma_bf16(qf0, load_frag(K + kof, Dc), s0);
    s0 = wmma_bf16(qf1, load_frag(K + kof + 32, Dc), s0);
    s1 = wmma_bf16(qf0, load_frag(K + kof + (size_t)16 * Dc, Dc), s1);
    s1 = wmma_bf16(qf1, load_frag(K + kof + (size_t)16 * Dc + 32, Dc), s1);

    float al[8];
#pragma unroll
    for (int r = 0; r < 8; ++r) {
      const float v0 = s0[r] * scale;
      const float v1 = s1[r] * scale;
      const float cm = rmax16(fmaxf(v0, v1));
      const float nm = fmaxf(mi[r], cm);
      al[r] = __expf(mi[r] - nm);
      const float p0 = __expf(v0 - nm);
      const float p1 = __expf(v1 - nm);
      li[r] = li[r] * al[r] + rsum16(p0 + p1);
      mi[r] = nm;
      P[(mb + r) * 32 + nl]      = (__bf16)p0;   // D-layout -> row-major LDS
      P[(mb + r) * 32 + 16 + nl] = (__bf16)p1;
    }
#pragma unroll
    for (int j = 0; j < 4; ++j)
#pragma unroll
      for (int r = 0; r < 8; ++r) co[j][r] *= al[r];

    // per-wave LDS RAW: cross-lane through LDS, drain DS pipe explicitly
    asm volatile("s_wait_dscnt 0" ::: "memory");
    const v16bf pf = load_frag(P, 32);           // P as A-fragment (16x32 keys)
#pragma unroll
    for (int j = 0; j < 4; ++j) {
      const __bf16* vb = V + (((size_t)b * Hc + h) * DHc + j * 16) * Sc + kk;
      co[j] = wmma_bf16(pf, load_frag(vb, Sc), co[j]);
    }
  }

#pragma unroll
  for (int j = 0; j < 4; ++j) {
#pragma unroll
    for (int r = 0; r < 8; ++r) {
      const float val = co[j][r] / li[r];
      C[((size_t)b * Sc + q0 + mb + r) * Dc + (size_t)h * DHc + j * 16 + nl] =
          (__bf16)val;
    }
  }
}

// ---------------------------------------------------------------- launch
extern "C" void kernel_launch(void* const* d_in, const int* in_sizes, int n_in,
                              void* d_out, int out_size, void* d_ws,
                              size_t ws_size, hipStream_t stream) {
  (void)in_sizes; (void)n_in; (void)out_size; (void)ws_size;
  const float* x  = (const float*)d_in[0];
  const float* wq = (const float*)d_in[1];
  const float* bq = (const float*)d_in[2];
  const float* wk = (const float*)d_in[3];
  const float* bk = (const float*)d_in[4];
  const float* wv = (const float*)d_in[5];
  const float* bv = (const float*)d_in[6];
  const float* wo = (const float*)d_in[7];
  const float* bo = (const float*)d_in[8];
  float* out = (float*)d_out;

  __bf16* w = (__bf16*)d_ws;
  size_t o = 0;
  __bf16* XB = w + o; o += (size_t)Mtok * Dc;
  __bf16* WQ = w + o; o += (size_t)Dc * Dc;
  __bf16* WK = w + o; o += (size_t)Dc * Dc;
  __bf16* WV = w + o; o += (size_t)Dc * Dc;
  __bf16* WO = w + o; o += (size_t)Dc * Dc;
  __bf16* QB = w + o; o += (size_t)Mtok * Dc;
  __bf16* KB = w + o; o += (size_t)Mtok * Dc;
  __bf16* VT = w + o; o += (size_t)Mtok * Dc;  // [B,H,DH,S]
  __bf16* CT = w + o; o += (size_t)Mtok * Dc;  // ~92.3 MB total

  const int nx4 = Mtok * Dc / 4;  // 2,097,152
  const int nw4 = Dc * Dc / 4;    //   262,144
  cvt_f32_bf16<<<nx4 / 256, 256, 0, stream>>>(x, XB, nx4);
  cvt_f32_bf16<<<nw4 / 256, 256, 0, stream>>>(wq, WQ, nw4);
  cvt_f32_bf16<<<nw4 / 256, 256, 0, stream>>>(wk, WK, nw4);
  cvt_f32_bf16<<<nw4 / 256, 256, 0, stream>>>(wv, WV, nw4);
  cvt_f32_bf16<<<nw4 / 256, 256, 0, stream>>>(wo, WO, nw4);

  dim3 gg(Dc / 128, Mtok / 128);  // (8, 64)
  gemm_wmma_k<0><<<gg, 256, 0, stream>>>(XB, WQ, bq, QB, nullptr);
  gemm_wmma_k<0><<<gg, 256, 0, stream>>>(XB, WK, bk, KB, nullptr);
  gemm_wmma_k<1><<<gg, 256, 0, stream>>>(XB, WV, bv, VT, nullptr);

  attn_kernel<<<(Bc * Hc * (Sc / 16)) / 8, 256, 0, stream>>>(QB, KB, VT, CT);

  gemm_wmma_k<2><<<gg, 256, 0, stream>>>(CT, WO, bo, nullptr, out);
}
